// MultiHeadAttention_53652731462425
// MI455X (gfx1250) — matrix operand order repaired and checked by reading, not compiled, not run on hardware
//
#include <hip/hip_runtime.h>
#include <math.h>

// ---------------------------------------------------------------------------
// MI455X (gfx1250) multi-head attention: bf16 WMMA + async LDS double-buffer.
// ---------------------------------------------------------------------------

typedef __attribute__((ext_vector_type(16))) __bf16 v16bf;
typedef __attribute__((ext_vector_type(8)))  __bf16 v8bf;
typedef __attribute__((ext_vector_type(8)))  float  v8f;

#define D_MODEL  1024
#define N_HEADS  16
#define HEAD_DIM 64
#define SEQ      2048
#define BATCH    4
#define ROWS     (BATCH * SEQ)   // 8192

// ---- CDNA5 async global->LDS copy (ASYNCcnt path) via inline asm ----------
// global_load_async_to_lds_b128: VDST = LDS byte address (wave-relative; the
// low 32 bits of a generic pointer into LDS), VADDR = 64-bit global address.

#if defined(__gfx1250__)
#define USE_ASYNC_LDS 1
#else
#define USE_ASYNC_LDS 0
#endif

__device__ __forceinline__ void cp_async_b128(__bf16* dst_lds, const __bf16* src_g) {
#if USE_ASYNC_LDS
  unsigned lds_addr = (unsigned)(unsigned long long)(void*)dst_lds;
  unsigned long long gaddr = (unsigned long long)src_g;
  asm volatile("global_load_async_to_lds_b128 %0, %1, off"
               :: "v"(lds_addr), "v"(gaddr)
               : "memory");
#else
  *(v8bf*)dst_lds = *(const v8bf*)src_g;
#endif
}

__device__ __forceinline__ void cp_async_wait() {
#if USE_ASYNC_LDS
  asm volatile("s_wait_asynccnt 0x0" ::: "memory");
#endif
}

// ---- WMMA helpers ----------------------------------------------------------

__device__ __forceinline__ v8f wmma_bf16(v16bf a, v16bf b, v8f c) {
  // (neg_a, A, neg_b, B, c_mod, C, reuse_a, reuse_b)
  return __builtin_amdgcn_wmma_f32_16x16x32_bf16(false, a, false, b,
                                                 (short)0, c, false, false);
}

// A fragment (16x32 row-major): lanes 0-15 (M=lane) hold K=0..7 & 16..23,
// lanes 16-31 hold K=8..15 & 24..31 (per ISA A layout).
__device__ __forceinline__ v16bf frag_a(const __bf16* row, int lane) {
  const int koff = (lane & 16) >> 1;            // 0 or 8
  v8bf lo = *(const v8bf*)(row + koff);
  v8bf hi = *(const v8bf*)(row + koff + 16);
  v16bf r;
#pragma unroll
  for (int i = 0; i < 8; ++i) { r[i] = lo[i]; r[i + 8] = hi[i]; }
  return r;
}

// B fragment (32x16): column n = lane&15 contiguous along K; lanes 0-15 hold
// K=0..7 & 16..23, lanes 16-31 hold K=8..15 & 24..31 (matches A layout).
__device__ __forceinline__ v16bf frag_b(const __bf16* col, int lane) {
  const int koff = (lane & 16) >> 1;            // 0 or 8
  v8bf lo = *(const v8bf*)(col + koff);
  v8bf hi = *(const v8bf*)(col + koff + 16);
  v16bf r;
#pragma unroll
  for (int i = 0; i < 8; ++i) { r[i] = lo[i]; r[i + 8] = hi[i]; }
  return r;
}

// ---- f32 -> bf16 conversion ------------------------------------------------

__global__ void cvt_bf16_kernel(const float* __restrict__ in,
                                __bf16* __restrict__ out, int n) {
  int i = blockIdx.x * blockDim.x + threadIdx.x;
  if (i < n) out[i] = (__bf16)in[i];
}

// ---- generic bf16 GEMM: C[M,N] = A[M,K] * B[K,N] + bias --------------------
// block tile 64x128, 128 threads = 4 waves (2x2), each wave 32x64 (2x4 WMMA),
// double-buffered LDS with async A-tile fill.

__device__ __forceinline__ void store_out(float* p, float v)  { *p = v; }
__device__ __forceinline__ void store_out(__bf16* p, float v) { *p = (__bf16)v; }

template <typename OutT>
__global__ __launch_bounds__(128)
void gemm_bf16_kernel(const __bf16* __restrict__ A, const __bf16* __restrict__ B,
                      const float* __restrict__ bias, OutT* __restrict__ C,
                      int M, int N, int K) {
  __shared__ __bf16 As[2][64][32];    // A tile, row-major          (2 x 4 KB)
  __shared__ __bf16 Bs[2][128][32];   // B^T tile: Bs[n][k]=B[k][n] (2 x 8 KB)

  const int tid  = threadIdx.x;
  const int wave = tid >> 5, lane = tid & 31;
  const int wm   = wave >> 1, wn = wave & 1;
  const int lm   = lane & 15, half = lane >> 4;
  const int m0   = blockIdx.y * 64, n0 = blockIdx.x * 128;

  auto load_tiles = [&](int k0, int bufi) {
    {   // A: 64 rows x 32 k; async 2 x b128 per thread
      int row = tid >> 1, seg = tid & 1;
      const __bf16* src = A + (size_t)(m0 + row) * K + k0 + seg * 16;
      cp_async_b128(&As[bufi][row][seg * 16], src);
      cp_async_b128(&As[bufi][row][seg * 16 + 8], src + 8);
    }
    {   // B: 32 k-rows x 128 n; contiguous reads, transposed LDS writes
      int k = tid >> 2, seg = tid & 3;
      const __bf16* src = B + (size_t)(k0 + k) * N + n0 + seg * 32;
#pragma unroll
      for (int j = 0; j < 32; ++j) Bs[bufi][seg * 32 + j][k] = src[j];
    }
  };

  v8f acc[2][4];
#pragma unroll
  for (int a = 0; a < 2; ++a)
#pragma unroll
    for (int b = 0; b < 4; ++b)
#pragma unroll
      for (int j = 0; j < 8; ++j) acc[a][b][j] = 0.f;

  load_tiles(0, 0);
  cp_async_wait();
  __syncthreads();

  int buf = 0;
  for (int k0 = 0; k0 < K; k0 += 32) {
    if (k0 + 32 < K) load_tiles(k0 + 32, buf ^ 1);   // overlap with compute

    v16bf a0 = frag_a(&As[buf][wm * 32 + lm][0], lane);
    v16bf a1 = frag_a(&As[buf][wm * 32 + 16 + lm][0], lane);
#pragma unroll
    for (int ni = 0; ni < 4; ++ni) {
      v16bf bf = frag_b(&Bs[buf][wn * 64 + ni * 16 + lm][0], lane);
      acc[0][ni] = wmma_bf16(a0, bf, acc[0][ni]);
      acc[1][ni] = wmma_bf16(a1, bf, acc[1][ni]);
    }

    cp_async_wait();
    __syncthreads();
    buf ^= 1;
  }

#pragma unroll
  for (int mi = 0; mi < 2; ++mi)
#pragma unroll
    for (int ni = 0; ni < 4; ++ni) {
      int col = n0 + wn * 64 + ni * 16 + lm;
      float bval = bias[col];
#pragma unroll
      for (int j = 0; j < 8; ++j) {
        int row = m0 + wm * 32 + mi * 16 + half * 8 + j;
        store_out(&C[(size_t)row * N + col], acc[mi][ni][j] + bval);
      }
    }
}

// ---- flash attention -------------------------------------------------------
// grid (SEQ/64, N_HEADS, BATCH), 128 threads. 64 query rows per block
// (16 per wave), KV tiles of 32, online softmax in f32, double-buffered
// K/V tiles with async K-row fill.

__global__ __launch_bounds__(128)
void flash_attn_kernel(const __bf16* __restrict__ Q, const __bf16* __restrict__ K,
                       const __bf16* __restrict__ V, __bf16* __restrict__ Y) {
  __shared__ __bf16 Ks[2][32][64];   // K tile rows (B-layout for Q*K^T), 2x4KB
  __shared__ __bf16 Vt[2][64][32];   // V tile transposed: Vt[d][kc],     2x4KB
  __shared__ __bf16 Ps[4][16][32];   // per-wave P tile (C->A relayout),    4KB

  const int tid  = threadIdx.x, wave = tid >> 5, lane = tid & 31;
  const int lm   = lane & 15, half = lane >> 4;
  const int q0   = blockIdx.x * 64;
  const int h    = blockIdx.y, b = blockIdx.z;
  const size_t base = ((size_t)b * SEQ) * D_MODEL + (size_t)h * HEAD_DIM;

  auto load_kv = [&](int kt, int bufi) {
    int r = tid >> 2, seg = tid & 3;
    const __bf16* ksrc = K + base + (size_t)(kt * 32 + r) * D_MODEL + seg * 16;
    cp_async_b128(&Ks[bufi][r][seg * 16], ksrc);
    cp_async_b128(&Ks[bufi][r][seg * 16 + 8], ksrc + 8);
    const __bf16* vsrc = V + base + (size_t)(kt * 32 + r) * D_MODEL + seg * 16;
#pragma unroll
    for (int j = 0; j < 16; ++j) Vt[bufi][seg * 16 + j][r] = vsrc[j];
  };

  // Q fragments for this wave's 16 rows, kept in registers (2 K-steps of 32)
  const __bf16* qrow = Q + base + (size_t)(q0 + wave * 16 + lm) * D_MODEL;
  v16bf qf0 = frag_a(qrow, lane);
  v16bf qf1 = frag_a(qrow + 32, lane);

  float mi[8], li[8];
  v8f o[4];
#pragma unroll
  for (int j = 0; j < 8; ++j) { mi[j] = -INFINITY; li[j] = 0.f; }
#pragma unroll
  for (int t = 0; t < 4; ++t)
#pragma unroll
    for (int j = 0; j < 8; ++j) o[t][j] = 0.f;

  const float scale = 0.125f;   // 1/sqrt(64)

  load_kv(0, 0);
  cp_async_wait();
  __syncthreads();

  int buf = 0;
  for (int kt = 0; kt < SEQ / 32; ++kt) {
    if (kt + 1 < SEQ / 32) load_kv(kt + 1, buf ^ 1);   // overlap with compute

    // S[16x32] = Q * K^T  (B fragment = K rows along d, contiguous)
    v8f s0, s1;
#pragma unroll
    for (int j = 0; j < 8; ++j) { s0[j] = 0.f; s1[j] = 0.f; }
    s0 = wmma_bf16(qf0, frag_b(&Ks[buf][lm][0], lane),       s0);
    s1 = wmma_bf16(qf0, frag_b(&Ks[buf][16 + lm][0], lane),  s1);
    s0 = wmma_bf16(qf1, frag_b(&Ks[buf][lm][32], lane),      s0);
    s1 = wmma_bf16(qf1, frag_b(&Ks[buf][16 + lm][32], lane), s1);

    // online softmax: row m = j + 8*half lives in one 16-lane half
    float p0[8], p1[8], ef[8];
#pragma unroll
    for (int j = 0; j < 8; ++j) {
      float a = s0[j] * scale, c = s1[j] * scale;
      float mx = fmaxf(a, c);
      mx = fmaxf(mx, __shfl_xor(mx, 1, 32));
      mx = fmaxf(mx, __shfl_xor(mx, 2, 32));
      mx = fmaxf(mx, __shfl_xor(mx, 4, 32));
      mx = fmaxf(mx, __shfl_xor(mx, 8, 32));
      float mnew = fmaxf(mi[j], mx);
      ef[j] = __expf(mi[j] - mnew);
      mi[j] = mnew;
      p0[j] = __expf(a - mnew);
      p1[j] = __expf(c - mnew);
      float rs = p0[j] + p1[j];
      rs += __shfl_xor(rs, 1, 32);
      rs += __shfl_xor(rs, 2, 32);
      rs += __shfl_xor(rs, 4, 32);
      rs += __shfl_xor(rs, 8, 32);
      li[j] = li[j] * ef[j] + rs;
    }
#pragma unroll
    for (int t = 0; t < 4; ++t)
#pragma unroll
      for (int j = 0; j < 8; ++j) o[t][j] *= ef[j];

    // reshape P (C-layout) -> A-layout via per-wave LDS tile (wave-local;
    // LDS ops are in-order within a wave, no barrier needed)
#pragma unroll
    for (int j = 0; j < 8; ++j) {
      Ps[wave][half * 8 + j][lm]      = (__bf16)p0[j];
      Ps[wave][half * 8 + j][16 + lm] = (__bf16)p1[j];
    }
    v16bf pf = frag_a(&Ps[wave][lm][0], lane);
#pragma unroll
    for (int t = 0; t < 4; ++t)
      o[t] = wmma_bf16(pf, frag_b(&Vt[buf][t * 16 + lm][0], lane), o[t]);

    cp_async_wait();
    __syncthreads();
    buf ^= 1;
  }

  // normalize and store merged-head output (same layout as Q/K/V)
#pragma unroll
  for (int j = 0; j < 8; ++j) li[j] = 1.f / li[j];
#pragma unroll
  for (int t = 0; t < 4; ++t)
#pragma unroll
    for (int j = 0; j < 8; ++j) {
      int row = q0 + wave * 16 + half * 8 + j;
      Y[base + (size_t)row * D_MODEL + t * 16 + lm] = (__bf16)(o[t][j] * li[j]);
    }
}

// ---- host-side launch ------------------------------------------------------

extern "C" void kernel_launch(void* const* d_in, const int* in_sizes, int n_in,
                              void* d_out, int out_size, void* d_ws, size_t ws_size,
                              hipStream_t stream) {
  (void)in_sizes; (void)n_in; (void)out_size; (void)ws_size;
  const float* x  = (const float*)d_in[0];
  const float* Wq = (const float*)d_in[1];
  const float* bq = (const float*)d_in[2];
  const float* Wk = (const float*)d_in[3];
  const float* bk = (const float*)d_in[4];
  const float* Wv = (const float*)d_in[5];
  const float* bv = (const float*)d_in[6];
  const float* Wo = (const float*)d_in[7];
  const float* bo = (const float*)d_in[8];
  float* out = (float*)d_out;

  char* ws = (char*)d_ws;
  const size_t NX = (size_t)ROWS * D_MODEL;     // 8,388,608 elements
  const size_t NW = (size_t)D_MODEL * D_MODEL;  // 1,048,576 elements
  __bf16* xb  = (__bf16*)(ws);                       // 16 MB
  __bf16* wqb = (__bf16*)(ws + NX * 2);              //  2 MB each
  __bf16* wkb = (__bf16*)(ws + NX * 2 + NW * 2);
  __bf16* wvb = (__bf16*)(ws + NX * 2 + NW * 4);
  __bf16* wob = (__bf16*)(ws + NX * 2 + NW * 6);
  __bf16* Qb  = (__bf16*)(ws + NX * 2 + NW * 8);     // 16 MB each
  __bf16* Kb  = (__bf16*)(ws + NX * 4 + NW * 8);
  __bf16* Vb  = (__bf16*)(ws + NX * 6 + NW * 8);
  __bf16* Yb  = (__bf16*)(ws + NX * 8 + NW * 8);     // total ~88 MB

  const int thr = 256;
  cvt_bf16_kernel<<<(int)((NX + thr - 1) / thr), thr, 0, stream>>>(x,  xb,  (int)NX);
  cvt_bf16_kernel<<<(int)((NW + thr - 1) / thr), thr, 0, stream>>>(Wq, wqb, (int)NW);
  cvt_bf16_kernel<<<(int)((NW + thr - 1) / thr), thr, 0, stream>>>(Wk, wkb, (int)NW);
  cvt_bf16_kernel<<<(int)((NW + thr - 1) / thr), thr, 0, stream>>>(Wv, wvb, (int)NW);
  cvt_bf16_kernel<<<(int)((NW + thr - 1) / thr), thr, 0, stream>>>(Wo, wob, (int)NW);

  dim3 ggrid(D_MODEL / 128, ROWS / 64);  // (8, 128)
  gemm_bf16_kernel<__bf16><<<ggrid, 128, 0, stream>>>(xb, wqb, bq, Qb, ROWS, D_MODEL, D_MODEL);
  gemm_bf16_kernel<__bf16><<<ggrid, 128, 0, stream>>>(xb, wkb, bk, Kb, ROWS, D_MODEL, D_MODEL);
  gemm_bf16_kernel<__bf16><<<ggrid, 128, 0, stream>>>(xb, wvb, bv, Vb, ROWS, D_MODEL, D_MODEL);

  dim3 agrid(SEQ / 64, N_HEADS, BATCH);  // (32, 16, 4)
  flash_attn_kernel<<<agrid, 128, 0, stream>>>(Qb, Kb, Vb, Yb);

  gemm_bf16_kernel<float><<<ggrid, 128, 0, stream>>>(Yb, wob, bo, out, ROWS, D_MODEL, D_MODEL);
}
